// SlidingWindowAttention_10428180594734
// MI455X (gfx1250) — matrix operand order repaired and checked by reading, not compiled, not run on hardware
//
#include <hip/hip_runtime.h>

// ---------------- problem constants ----------------
#define B_   2
#define T_   2048
#define D_   1024
#define H_   16
#define DH_  64
#define WIN_ 256
#define E3_  3072
#define SCALE_ 0.125f   // 64^-0.5

typedef __attribute__((ext_vector_type(16))) __bf16 v16bf;
typedef __attribute__((ext_vector_type(8)))  float  v8f;

union BF16x16 { v16bf v; unsigned u[8]; };

__device__ __forceinline__ unsigned short f2bf(float f) {
    unsigned x = __float_as_uint(f);
    unsigned r = x + 0x7fffu + ((x >> 16) & 1u);   // round-to-nearest-even
    return (unsigned short)(r >> 16);
}
__device__ __forceinline__ unsigned pack2(float lo, float hi) {
    return (unsigned)f2bf(lo) | ((unsigned)f2bf(hi) << 16);
}

// async DMA: global -> LDS, 16B per lane, tracked by ASYNCcnt
#define ASYNC_LDS_B128(dst, src, off)                                       \
    asm volatile("global_load_async_to_lds_b128 %0, %1, off offset:" #off   \
                 :: "v"(dst), "v"(src) : "memory")
#define WAIT_ASYNC() asm volatile("s_wait_asynccnt 0x0" ::: "memory")
#define WAIT_DS()    asm volatile("s_wait_dscnt 0x0" ::: "memory")

// =====================================================================
// GEMM: C[m,n] = sum_k A[m,k] * W[n,k]   (A:[M,K] f32, W:[N,K] f32)
// Block: 128 thr = 4 waves; tile 64(M) x 64(N); K stepped by 32.
// Wave w owns rows [m0+16w, m0+16w+16) x all 64 N -> 4 WMMAs / K-step.
// W is staged transposed in LDS ([k][n]); B fragments are gathered as u16 pairs.
// MODE 0: QKV epilogue -> scatter bf16 into per-head Q/K/V workspace
// MODE 1: plain fp32 store to outF
// =====================================================================
template <int MODE>
__global__ __launch_bounds__(128) void gemm_bf16(
    const float* __restrict__ A, const float* __restrict__ W,
    int M, int N, int K,
    float* __restrict__ outF,
    unsigned short* __restrict__ qws,
    unsigned short* __restrict__ kws,
    unsigned short* __restrict__ vws)
{
    // A natural [m(64)][k-pair(16)+pad4]  (uint rows of 20 -> 80B, 16B aligned)
    __shared__ unsigned lds_a[64 * 20];
    // W^T [k(32)][n(64)+pad16] u16 rows of 80 -> 160B, 16B aligned
    __shared__ unsigned short lds_wt[32 * 80];

    const int tid  = threadIdx.x;
    const int lane = tid & 31;
    const int wave = tid >> 5;
    const int half = lane >> 4;
    const int l16  = lane & 15;
    const int m0   = blockIdx.x * 64;
    const int n0   = blockIdx.y * 64;

    v8f acc[4];
    #pragma unroll
    for (int nb = 0; nb < 4; ++nb) acc[nb] = (v8f){0.f,0.f,0.f,0.f,0.f,0.f,0.f,0.f};

    const unsigned* wt_u = (const unsigned*)lds_wt;

    for (int k0 = 0; k0 < K; k0 += 32) {
        __syncthreads();
        // stage A tile: 1024 uints (64 m x 16 k-pairs), coalesced along k
        #pragma unroll 8
        for (int u = tid; u < 1024; u += 128) {
            int m = u >> 4, kp = u & 15;
            const float* s = &A[(size_t)(m0 + m) * K + k0 + kp * 2];
            lds_a[m * 20 + kp] = pack2(s[0], s[1]);
        }
        // stage W^T: read coalesced along k, scatter u16 transposed
        #pragma unroll 8
        for (int u = tid; u < 1024; u += 128) {
            int n = u >> 4, kp = u & 15;
            const float* s = &W[(size_t)(n0 + n) * K + k0 + kp * 2];
            lds_wt[(2 * kp) * 80 + n]     = f2bf(s[0]);
            lds_wt[(2 * kp + 1) * 80 + n] = f2bf(s[1]);
        }
        __syncthreads();

        // A fragment: lane <-> M row, VGPR v <-> k-pair per ISA kmap
        BF16x16 a;
        const int row = wave * 16 + l16;
        #pragma unroll
        for (int v = 0; v < 8; ++v) {
            int k2 = ((v & 4) ? 8 : 0) + (v & 3) + half * 4;
            a.u[v] = lds_a[row * 20 + k2];
        }
        // B fragments: lane <-> n col of W^T, VGPR v <-> k-pair per ISA kmap
        #pragma unroll
        for (int nb = 0; nb < 4; ++nb) {
            BF16x16 b;
            #pragma unroll
            for (int v = 0; v < 8; ++v) {
                int kk = 2 * (((v & 4) ? 8 : 0) + (v & 3) + half * 4);
                int n = nb * 16 + l16;
                b.u[v] = (unsigned)lds_wt[kk * 80 + n] | ((unsigned)lds_wt[(kk + 1) * 80 + n] << 16);
            }
            acc[nb] = __builtin_amdgcn_wmma_f32_16x16x32_bf16(
                false, a.v, false, b.v, (short)0, acc[nb], false, false);
        }
    }

    // epilogue: D layout — lane <-> N col, VGPR r <-> M row (+8*half)
    #pragma unroll
    for (int nb = 0; nb < 4; ++nb) {
        #pragma unroll
        for (int r = 0; r < 8; ++r) {
            int m   = m0 + wave * 16 + r + 8 * half;
            int col = n0 + nb * 16 + l16;
            float val = acc[nb][r];
            if (MODE == 0) {
                int which = col >> 10;           // 0=q 1=k 2=v
                int rem   = col & 1023;
                int h     = rem >> 6, dh = rem & 63;
                int bb    = m / T_, t = m % T_;
                size_t o  = (((size_t)(bb * H_ + h) * T_) + t) * DH_ + dh;
                unsigned short bf = f2bf(val);
                if (which == 0)      qws[o] = bf;
                else if (which == 1) kws[o] = bf;
                else                 vws[o] = bf;
            } else {
                outF[(size_t)m * N + col] = val;
            }
        }
    }
}

// =====================================================================
// Flash attention over sliding causal window.
// Block: 128 thr = 4 waves; wave w owns query tile qt0 = bx*64 + w*16.
// Each wave has a private LDS slice (no inter-wave barriers).
// Q and V tiles are fetched with GLOBAL_LOAD_ASYNC_TO_LDS_B128 (ASYNCcnt);
// K is transposed into LDS through VGPRs.
// =====================================================================
__global__ __launch_bounds__(128) void attn_kernel(
    const unsigned short* __restrict__ Q,
    const unsigned short* __restrict__ Kb,
    const unsigned short* __restrict__ Vb,
    float* __restrict__ O)
{
    // per-wave u16 slice (all row strides 16B-aligned, non-pow2 for banks):
    //   lq : 16 rows x 72  (Q 16x64 + pad)        = 1152
    //   lkt: 64 rows x 40  (K^T 64x32 + pad)      = 2560
    //   lv : 32 rows x 72  (V 32x64 + pad)        = 2304
    //   lp : 16 rows x 40  (P 16x32 + pad)        =  640
    __shared__ unsigned short smem[4 * 6656];

    const int tid  = threadIdx.x;
    const int lane = tid & 31;
    const int wave = tid >> 5;
    unsigned short* lq  = &smem[wave * 6656];
    unsigned short* lkt = lq + 1152;
    unsigned short* lv  = lkt + 2560;
    unsigned short* lp  = lv + 2304;
    unsigned* lq_u  = (unsigned*)lq;
    unsigned* lkt_u = (unsigned*)lkt;
    unsigned* lv_u  = (unsigned*)lv;
    unsigned* lp_u  = (unsigned*)lp;
    const unsigned lq_addr = (unsigned)(size_t)lq;
    const unsigned lv_addr = (unsigned)(size_t)lv;

    const int b   = blockIdx.z;
    const int h   = blockIdx.y;
    const int qt0 = blockIdx.x * 64 + wave * 16;
    const size_t headbase = ((size_t)(b * H_ + h)) * T_ * DH_;

    const int half = lane >> 4;
    const int l16  = lane & 15;

    // ---- async DMA Q tile (16 rows x 128B) into LDS (rows padded to 144B) ----
    {
        int qrow = lane >> 1;                 // 2 lanes per row
        int hoff = (lane & 1) * 32;           // u16 offset within row (64B)
        const unsigned short* qs = Q + headbase + (size_t)(qt0 + qrow) * DH_ + hoff;
        unsigned qd = lq_addr + (unsigned)(qrow * 144 + hoff * 2);
        ASYNC_LDS_B128(qd, qs, 0);
        ASYNC_LDS_B128(qd, qs, 16);
        ASYNC_LDS_B128(qd, qs, 32);
        ASYNC_LDS_B128(qd, qs, 48);
    }
    WAIT_ASYNC();

    // A fragments of Q for dh halves [0,32) and [32,64)  (uint row stride 36)
    BF16x16 aQ0, aQ1;
    #pragma unroll
    for (int v = 0; v < 8; ++v) {
        int k2 = ((v & 4) ? 8 : 0) + (v & 3) + half * 4;
        aQ0.u[v] = lq_u[l16 * 36 + k2];
        aQ1.u[v] = lq_u[l16 * 36 + 16 + k2];
    }

    float mi[8], li[8];
    v8f acc[4];
    #pragma unroll
    for (int r = 0; r < 8; ++r) { mi[r] = -1e30f; li[r] = 0.f; }
    #pragma unroll
    for (int nb = 0; nb < 4; ++nb) acc[nb] = (v8f){0.f,0.f,0.f,0.f,0.f,0.f,0.f,0.f};

    int cstart = qt0 - (WIN_ - 1);
    if (cstart < 0) cstart = 0;
    cstart &= ~31;

    for (int c0 = cstart; c0 <= qt0 + 15; c0 += 32) {
        // make sure prior DS reads of lv/lkt/lp finished before overwriting
        WAIT_DS();

        // ---- async DMA V chunk: 32 rows x 128B -> LDS rows padded to 144B ----
        {
            int vrow = c0 + lane;
            if (vrow > T_ - 1) vrow = T_ - 1;       // clamped rows are masked (p=0)
            const unsigned short* vs = Vb + headbase + (size_t)vrow * DH_;
            unsigned vd = lv_addr + (unsigned)(lane * 144);
            ASYNC_LDS_B128(vd, vs, 0);
            ASYNC_LDS_B128(vd, vs, 16);
            ASYNC_LDS_B128(vd, vs, 32);
            ASYNC_LDS_B128(vd, vs, 48);
            ASYNC_LDS_B128(vd, vs, 64);
            ASYNC_LDS_B128(vd, vs, 80);
            ASYNC_LDS_B128(vd, vs, 96);
            ASYNC_LDS_B128(vd, vs, 112);
        }

        // ---- K chunk -> LDS transposed [dh][key] (u16 rows of 40) ----
        const unsigned* kgu = (const unsigned*)(Kb + headbase + (size_t)c0 * DH_);
        const unsigned* kgu_last =
            (const unsigned*)(Kb + headbase + (size_t)(T_ - 32) * DH_);
        #pragma unroll 4
        for (int t = 0; t < 32; ++t) {
            int iu  = lane + 32 * t;          // uint idx in [0,1024)
            int key = iu >> 5;
            int dh2 = (iu & 31) * 2;
            // clamp whole-chunk overrun (masked keys -> p = 0 anyway)
            unsigned d = (c0 + 31 < T_) ? kgu[iu] : kgu_last[iu];
            lkt[dh2 * 40 + key]       = (unsigned short)(d & 0xffffu);
            lkt[(dh2 + 1) * 40 + key] = (unsigned short)(d >> 16);
        }
        WAIT_DS();
        WAIT_ASYNC();

        // ---- scores S = Q * K^T for two 16-key subtiles ----
        v8f S[2];
        #pragma unroll
        for (int j = 0; j < 2; ++j) {
            BF16x16 b0, b1;   // B = K^T: lane <-> key(n), VGPR v <-> dh pair per ISA kmap
            #pragma unroll
            for (int v = 0; v < 8; ++v) {
                int kk = 2 * (((v & 4) ? 8 : 0) + (v & 3) + half * 4);
                int n = j * 16 + l16;
                b0.u[v] = (unsigned)lkt[kk * 40 + n] | ((unsigned)lkt[(kk + 1) * 40 + n] << 16);
                b1.u[v] = (unsigned)lkt[(kk + 32) * 40 + n] | ((unsigned)lkt[(kk + 33) * 40 + n] << 16);
            }
            v8f s = (v8f){0.f,0.f,0.f,0.f,0.f,0.f,0.f,0.f};
            s = __builtin_amdgcn_wmma_f32_16x16x32_bf16(false, aQ0.v, false, b0.v,
                                                        (short)0, s, false, false);
            s = __builtin_amdgcn_wmma_f32_16x16x32_bf16(false, aQ1.v, false, b1.v,
                                                        (short)0, s, false, false);
            S[j] = s;
        }

        // ---- mask + online softmax (row groups = 16 lanes; xor<=8 stays inside) ----
        float p0a[8], p1a[8];
        #pragma unroll
        for (int r = 0; r < 8; ++r) {
            int q   = qt0 + r + 8 * half;
            int k0i = c0 + l16;
            int k1i = k0i + 16;
            float s0 = S[0][r] * SCALE_;
            float s1 = S[1][r] * SCALE_;
            bool a0 = (q >= k0i) && (q - k0i < WIN_);
            bool a1 = (q >= k1i) && (q - k1i < WIN_);
            s0 = a0 ? s0 : -1e30f;
            s1 = a1 ? s1 : -1e30f;
            float mx = fmaxf(s0, s1);
            #pragma unroll
            for (int d = 1; d < 16; d <<= 1) mx = fmaxf(mx, __shfl_xor(mx, d, 32));
            float mn    = fmaxf(mi[r], mx);
            float alpha = __expf(mi[r] - mn);          // both -1e30 -> exp(0)=1
            float p0 = (s0 < -1e29f) ? 0.f : __expf(s0 - mn);
            float p1 = (s1 < -1e29f) ? 0.f : __expf(s1 - mn);
            float rs = p0 + p1;
            #pragma unroll
            for (int d = 1; d < 16; d <<= 1) rs += __shfl_xor(rs, d, 32);
            li[r] = li[r] * alpha + rs;
            mi[r] = mn;
            #pragma unroll
            for (int nb = 0; nb < 4; ++nb) acc[nb][r] *= alpha;
            p0a[r] = p0; p1a[r] = p1;
        }

        // ---- P (D layout) -> LDS bf16 [16 m][32 key] (u16 rows of 40) ----
        #pragma unroll
        for (int r = 0; r < 8; ++r) {
            int m = r + 8 * half;
            lp[m * 40 + l16]      = f2bf(p0a[r]);
            lp[m * 40 + 16 + l16] = f2bf(p1a[r]);
        }
        WAIT_DS();

        // ---- A fragment of P (16x32), uint row stride 20 ----
        BF16x16 aP;
        #pragma unroll
        for (int v = 0; v < 8; ++v) {
            int k2 = ((v & 4) ? 8 : 0) + (v & 3) + half * 4;
            aP.u[v] = lp_u[l16 * 20 + k2];
        }
        // ---- acc += P * V  (B = V: lane <-> dh, VGPR v <-> key pair per ISA kmap) ----
        #pragma unroll
        for (int nb = 0; nb < 4; ++nb) {
            BF16x16 bv;
            #pragma unroll
            for (int v = 0; v < 8; ++v) {
                int kk = 2 * (((v & 4) ? 8 : 0) + (v & 3) + half * 4);
                int n = nb * 16 + l16;
                bv.u[v] = (unsigned)lv[kk * 72 + n] | ((unsigned)lv[(kk + 1) * 72 + n] << 16);
            }
            acc[nb] = __builtin_amdgcn_wmma_f32_16x16x32_bf16(false, aP.v, false, bv.v,
                                                              (short)0, acc[nb], false, false);
        }
    }

    // ---- epilogue: O[b, t, h*64 + dh] = acc / l ----
    #pragma unroll
    for (int nb = 0; nb < 4; ++nb) {
        #pragma unroll
        for (int r = 0; r < 8; ++r) {
            int t   = qt0 + r + 8 * half;
            int col = h * DH_ + nb * 16 + l16;
            O[((size_t)b * T_ + t) * (H_ * DH_) + col] = acc[nb][r] / li[r];
        }
    }
}

// =====================================================================
// Launcher.  Workspace layout:
//   [0,  8M)  Q bf16  [B,H,T,64]
//   [8M, 16M) K bf16
//   [16M,24M) V bf16
//   [24M,40M) O fp32  [B,T,1024]
// =====================================================================
extern "C" void kernel_launch(void* const* d_in, const int* in_sizes, int n_in,
                              void* d_out, int out_size, void* d_ws, size_t ws_size,
                              hipStream_t stream) {
    const float* x     = (const float*)d_in[0];
    const float* w_qkv = (const float*)d_in[1];
    const float* w_out = (const float*)d_in[2];
    float* out = (float*)d_out;

    const size_t NHEADEL = (size_t)B_ * H_ * T_ * DH_;   // 4,194,304
    unsigned short* qws = (unsigned short*)d_ws;
    unsigned short* kws = qws + NHEADEL;
    unsigned short* vws = kws + NHEADEL;
    float* Ows = (float*)(vws + NHEADEL);

    const int M = B_ * T_;   // 4096

    // 1) QKV projection -> per-head bf16 Q/K/V
    dim3 g1(M / 64, E3_ / 64);
    gemm_bf16<0><<<g1, 128, 0, stream>>>(x, w_qkv, M, E3_, D_,
                                         nullptr, qws, kws, vws);

    // 2) sliding-window flash attention -> O fp32 [B,T,1024]
    dim3 g2(T_ / 64, H_, B_);
    attn_kernel<<<g2, 128, 0, stream>>>(qws, kws, vws, Ows);

    // 3) output projection -> d_out
    dim3 g3(M / 64, D_ / 64);
    gemm_bf16<1><<<g3, 128, 0, stream>>>(Ows, w_out, M, D_, D_,
                                         out, nullptr, nullptr, nullptr);
}
